// NWKernelRegression_44882408243250
// MI455X (gfx1250) — matrix operand order repaired and checked by reading, not compiled, not run on hardware
//
#include <hip/hip_runtime.h>

// NW kernel regression on MI455X (gfx1250, wave32).
// out[i] = sum_j exp(-0.5*((q_i-x_j)*w_j)^2) * y_j / sum_j exp(...)
// Scores are <= 0 so exp is in (0,1]: single pass, no max-subtraction needed.
//
// Per 16-query x 32-train tile:
//   VALU:  t = fma(q, w*r, -(x*w*r));  u = exp2(-(t*t))   [v_exp_f32]
//   WMMA:  D(16x16,f32) += A(16x32,f16:[y; 1; 0...]) x B(32x16,f16: exp tile)
//   -> D[0,n] = numerator(query n), D[1,n] = denominator(query n)

typedef __attribute__((ext_vector_type(16))) _Float16 v16h;
typedef __attribute__((ext_vector_type(8)))  _Float16 v8h;
typedef __attribute__((ext_vector_type(8)))  float    v8f;
typedef __attribute__((ext_vector_type(4)))  float    v4f;

#define N_TRAIN 16384
#define WAVES_PER_BLOCK 8
#define BLOCK_THREADS (WAVES_PER_BLOCK * 32)

__global__ __launch_bounds__(BLOCK_THREADS) void nw_wmma_kernel(
    const float* __restrict__ queries, const float* __restrict__ x_train,
    const float* __restrict__ y_train, const float* __restrict__ w,
    float* __restrict__ out) {
  // LDS: A1 = w*r (f32), A2 = x*w*r (f32), Y = y (f16).  160 KB total.
  extern __shared__ char smem[];
  float*    sA1 = (float*)smem;
  float*    sA2 = sA1 + N_TRAIN;
  _Float16* sY  = (_Float16*)(sA2 + N_TRAIN);

  // r = sqrt(log2(e)/2): score*log2(e) = -(r*d*w)^2, so we can use v_exp_f32.
  const float r = 0.84932180028802f;

  for (int j = threadIdx.x; j < N_TRAIN; j += BLOCK_THREADS) {
    float wj = w[j];
    sA1[j] = wj * r;
    sA2[j] = x_train[j] * wj * r;
    sY[j]  = (_Float16)y_train[j];
  }
  __syncthreads();

  const int  lane  = threadIdx.x & 31;
  const int  wave  = threadIdx.x >> 5;
  const int  qtile = blockIdx.x * WAVES_PER_BLOCK + wave;  // 16 queries per wave
  const int  rrow  = lane & 15;
  const bool hi    = lane >= 16;
  const int  joff  = hi ? 16 : 0;   // B: lanes 16-31 hold K=16..31
  const int  ybase = hi ? 16 : 0;   // A: lanes 16-31 hold K=16..31

  const float qv = queries[qtile * 16 + rrow];  // this lane's query (N = lane&15)

  v16h ones16;
#pragma unroll
  for (int e = 0; e < 16; ++e) ones16[e] = (_Float16)1.0f;

  v8f acc = {};  // D accumulator: row0 = numerator, row1 = denominator

#pragma unroll 2
  for (int jb = 0; jb < N_TRAIN; jb += 32) {
    // ---- B tile: exp scores for this lane's query, j = jb + joff + e ----
    const v4f* pa1 = (const v4f*)(sA1 + jb + joff);  // broadcast LDS reads
    const v4f* pa2 = (const v4f*)(sA2 + jb + joff);
    v16h b;
#pragma unroll
    for (int g = 0; g < 4; ++g) {
      v4f a1 = pa1[g];
      v4f a2 = pa2[g];
#pragma unroll
      for (int h = 0; h < 4; ++h) {
        float t = __builtin_fmaf(qv, a1[h], -a2[h]);   // r*(q - x_j)*w_j
        float u = __builtin_amdgcn_exp2f(-(t * t));    // exp(-0.5*(d*w)^2)
        b[g * 4 + h] = (_Float16)u;
      }
    }

    // ---- A tile: M=0 row = y_j, M=1 row = 1.0, other rows = 0 ----
    // Lane (&15)==0 carries row 0: elements 0..7 <- K=ybase+0..7,
    //                              elements 8..15 <- K=ybase+8..15.
    v16h a = {};
    if (rrow == 0) {
      const v8h* py = (const v8h*)(sY + jb + ybase);
      v8h ylo = py[0];  // y[jb+ybase   .. +7 ]
      v8h yhi = py[1];  // y[jb+ybase+8 .. +15]
#pragma unroll
      for (int e = 0; e < 8; ++e) { a[e] = ylo[e]; a[8 + e] = yhi[e]; }
    } else if (rrow == 1) {
      a = ones16;
    }

    // D = A x B + C  -> v_wmma_f32_16x16x32_f16
    acc = __builtin_amdgcn_wmma_f32_16x16x32_f16(
        /*neg_a=*/false, a, /*neg_b=*/false, b,
        /*c_mod=*/(short)0, acc, /*reuse_a=*/false, /*reuse_b=*/false);
  }

  // D layout: VGPR0 = M=0 (numerator), VGPR1 = M=1 (denominator), N = lane (0..15).
  if (!hi) {
    out[qtile * 16 + rrow] = acc[0] / acc[1];
  }
}

extern "C" void kernel_launch(void* const* d_in, const int* in_sizes, int n_in,
                              void* d_out, int out_size, void* d_ws, size_t ws_size,
                              hipStream_t stream) {
  const float* queries = (const float*)d_in[0];
  const float* x_train = (const float*)d_in[1];
  const float* y_train = (const float*)d_in[2];
  const float* w       = (const float*)d_in[3];
  float* out = (float*)d_out;

  const int n_queries = in_sizes[0];                       // 16384
  const int blocks = n_queries / (16 * WAVES_PER_BLOCK);   // 128
  const size_t shmem = (size_t)N_TRAIN * (4 + 4 + 2);      // A1,A2 f32 + Y f16 = 160 KB

  nw_wmma_kernel<<<blocks, BLOCK_THREADS, shmem, stream>>>(
      queries, x_train, y_train, w, out);
}
